// InsZeroShotClassifier_3332894622298
// MI455X (gfx1250) — matrix operand order repaired and checked by reading, not compiled, not run on hardware
//
#include <hip/hip_runtime.h>
#include <math.h>

// ---------------------------------------------------------------------------
// CDNA5 async-to-LDS support (probe-confirmed present on this toolchain).
// Prototype (from compiler diagnostic): 
//   void __builtin_amdgcn_global_load_async_to_lds_b128(
//        v4i AS1* src, v4i AS3* dst, imm int offset, imm int cpol)
// ---------------------------------------------------------------------------
#if defined(__has_builtin) && defined(__AMDGCN__)
#if __has_builtin(__builtin_amdgcn_global_load_async_to_lds_b128) && \
    __has_builtin(__builtin_amdgcn_s_wait_asynccnt)
#define HAS_ASYNC_LDS 1
#endif
#endif
#ifndef HAS_ASYNC_LDS
#define HAS_ASYNC_LDS 0
#endif

typedef __bf16 v16bf __attribute__((ext_vector_type(16)));
typedef float  v8f   __attribute__((ext_vector_type(8)));
typedef unsigned int v4u __attribute__((ext_vector_type(4)));
typedef int v4i_gcc __attribute__((vector_size(16)));

union FragBF { v16bf v; v4u q[2]; };

__device__ __forceinline__ unsigned short f2bf(float f) {
  unsigned int u = __float_as_uint(f);
  unsigned int r = u + 0x7FFFu + ((u >> 16) & 1u);   // round-to-nearest-even
  return (unsigned short)(r >> 16);
}
__device__ __forceinline__ float bf2f(unsigned short u) {
  return __uint_as_float(((unsigned int)u) << 16);
}

#if HAS_ASYNC_LDS
// 16-byte async global->LDS copy. AS1 value == flat VA for global memory;
// AS3 value == flat VA truncated to 32 bits (ISA 10.2).
__device__ __forceinline__ void async_copy16(const unsigned short* g,
                                             unsigned short* l) {
  __builtin_amdgcn_global_load_async_to_lds_b128(
      (__attribute__((address_space(1))) v4i_gcc*)(unsigned long long)(const void*)g,
      (__attribute__((address_space(3))) v4i_gcc*)(unsigned int)(unsigned long long)(const void*)l,
      0, 0);
}
#endif

// ---------------------------------------------------------------------------
// Weight prep: f32 -> bf16 (transposed for GEMM B operands).
// ---------------------------------------------------------------------------
__global__ void convert_bf16_kernel(const float* __restrict__ in,
                                    unsigned short* __restrict__ out, int n)
{
  int idx = blockIdx.x * 256 + threadIdx.x;
  if (idx < n) out[idx] = f2bf(in[idx]);
}

// W[K][N] (f32, row-major) -> Wt[N][K] (bf16)
__global__ void transpose_bf16_kernel(const float* __restrict__ W,
                                      unsigned short* __restrict__ Wt,
                                      int K, int N)
{
  int idx = blockIdx.x * 256 + threadIdx.x;
  if (idx >= K * N) return;
  int n = idx / K, k = idx % K;
  Wt[idx] = f2bf(W[(size_t)k * N + n]);
}

// ---------------------------------------------------------------------------
// Stage one 64x32 A tile + 64x32 B tile (bf16, 16B chunks) into LDS.
// ---------------------------------------------------------------------------
__device__ __forceinline__ void stage_tile(const unsigned short* __restrict__ A,
                                           const unsigned short* __restrict__ Bt,
                                           unsigned short* As, unsigned short* Bs,
                                           int tid, int m0, int n0,
                                           int M, int N, int K, int k0)
{
#pragma unroll
  for (int j = 0; j < 2; ++j) {
    int q  = tid + 128 * j;            // 0..255 chunk id
    int r  = q >> 2;                   // tile row 0..63
    int c8 = q & 3;                    // 8-element chunk in k
    int gm = m0 + r; if (gm > M - 1) gm = M - 1;
    int gn = n0 + r; if (gn > N - 1) gn = N - 1;
#if HAS_ASYNC_LDS
    async_copy16(&A[(size_t)gm * K + k0 + c8 * 8],  &As[r * 40 + c8 * 8]);
    async_copy16(&Bt[(size_t)gn * K + k0 + c8 * 8], &Bs[r * 40 + c8 * 8]);
#else
    *reinterpret_cast<v4u*>(&As[r * 40 + c8 * 8]) =
        *reinterpret_cast<const v4u*>(&A[(size_t)gm * K + k0 + c8 * 8]);
    *reinterpret_cast<v4u*>(&Bs[r * 40 + c8 * 8]) =
        *reinterpret_cast<const v4u*>(&Bt[(size_t)gn * K + k0 + c8 * 8]);
#endif
  }
}

// ---------------------------------------------------------------------------
// GEMM: C[M,N] = epilogue(A[M,K](bf16) @ Bt[N,K](bf16)^T + bias)
// EP: 0 = bias, 1 = QuickGELU, 2 = bias + f32 residual.  OBF: 1 = bf16 out.
// Double-buffered async global->LDS pipeline on ASYNCcnt; WMMA bf16 core.
// ---------------------------------------------------------------------------
template<int EP, int OBF>
__global__ __launch_bounds__(128)
void gemm_kernel(const unsigned short* __restrict__ A,
                 const unsigned short* __restrict__ Bt,
                 const float* __restrict__ bias, const float* __restrict__ Res,
                 void* __restrict__ Cv, int M, int N, int K)
{
  __shared__ __align__(16) unsigned short As[2][64 * 40];  // [m][k] k-stride 40
  __shared__ __align__(16) unsigned short Bs[2][64 * 40];  // [n][k] k-stride 40

  const int tid  = threadIdx.x;
  const int wave = tid >> 5;
  const int lane = tid & 31;
  const int half = lane >> 4;
  const int l16  = lane & 15;
  const int wr   = wave >> 1;
  const int wc   = wave & 1;
  const int m0   = blockIdx.y * 64;
  const int n0   = blockIdx.x * 64;

  v8f acc00 = {}; v8f acc01 = {}; v8f acc10 = {}; v8f acc11 = {};

  const int nt = K >> 5;
  stage_tile(A, Bt, As[0], Bs[0], tid, m0, n0, M, N, K, 0);
  for (int t = 0; t < nt; ++t) {
    const int cur = t & 1;
    if (t + 1 < nt) {
      stage_tile(A, Bt, As[(t + 1) & 1], Bs[(t + 1) & 1], tid, m0, n0, M, N, K,
                 (t + 1) * 32);
#if HAS_ASYNC_LDS
      __builtin_amdgcn_s_wait_asynccnt(4);   // tile t retired, t+1 in flight
#endif
    } else {
#if HAS_ASYNC_LDS
      __builtin_amdgcn_s_wait_asynccnt(0);
#endif
    }
    __syncthreads();

    const unsigned short* Ac = As[cur];
    const unsigned short* Bc = Bs[cur];
    FragBF a0, a1, b0, b1;
    {
      int mA = wr * 32 + l16;
      const v4u* p0 = reinterpret_cast<const v4u*>(&Ac[mA * 40 + half * 8]);
      a0.q[0] = p0[0]; a0.q[1] = p0[2];
      const v4u* p1 = reinterpret_cast<const v4u*>(&Ac[(mA + 16) * 40 + half * 8]);
      a1.q[0] = p1[0]; a1.q[1] = p1[2];
      int nB = wc * 32 + l16;
      const v4u* pb0 = reinterpret_cast<const v4u*>(&Bc[nB * 40 + half * 8]);
      b0.q[0] = pb0[0]; b0.q[1] = pb0[2];
      const v4u* pb1 = reinterpret_cast<const v4u*>(&Bc[(nB + 16) * 40 + half * 8]);
      b1.q[0] = pb1[0]; b1.q[1] = pb1[2];
    }
    acc00 = __builtin_amdgcn_wmma_f32_16x16x32_bf16(false, a0.v, false, b0.v, (short)0, acc00, false, false);
    acc01 = __builtin_amdgcn_wmma_f32_16x16x32_bf16(false, a0.v, false, b1.v, (short)0, acc01, false, false);
    acc10 = __builtin_amdgcn_wmma_f32_16x16x32_bf16(false, a1.v, false, b0.v, (short)0, acc10, false, false);
    acc11 = __builtin_amdgcn_wmma_f32_16x16x32_bf16(false, a1.v, false, b1.v, (short)0, acc11, false, false);
    __syncthreads();
  }

  const int mw = m0 + wr * 32;
  const int nw = n0 + wc * 32;
  const bool interior = (m0 + 64 <= M) && (n0 + 64 <= N);
#pragma unroll
  for (int ci = 0; ci < 2; ++ci) {
#pragma unroll
    for (int cj = 0; cj < 2; ++cj) {
      v8f a = (ci == 0) ? (cj == 0 ? acc00 : acc01) : (cj == 0 ? acc10 : acc11);
      int col = nw + cj * 16 + l16;
      if (interior || col < N) {
        float bv = bias ? bias[col] : 0.0f;
#pragma unroll
        for (int r = 0; r < 8; ++r) {
          int row = mw + ci * 16 + half * 8 + r;
          if (interior || row < M) {
            float v = a[r] + bv;
            if (EP == 1) v = v / (1.0f + __expf(-1.702f * v));
            if (EP == 2) v += Res[(size_t)row * N + col];
            if (OBF) ((unsigned short*)Cv)[(size_t)row * N + col] = f2bf(v);
            else     ((float*)Cv)[(size_t)row * N + col] = v;
          }
        }
      }
    }
  }
}

// ---------------------------------------------------------------------------
// WMMA causal attention, L=77, dh=64, H=8. bf16 in (qkv), bf16 out.
// ---------------------------------------------------------------------------
__global__ __launch_bounds__(160)
void attn77_wmma_kernel(const unsigned short* __restrict__ qkv,
                        unsigned short* __restrict__ out)
{
  __shared__ __align__(16) unsigned short Qs[80 * 64];
  __shared__ __align__(16) unsigned short Ksh[80 * 64];
  __shared__ __align__(16) unsigned short Vt[64 * 96];
  __shared__ __align__(16) unsigned short Ps[80 * 96];
  const int b = blockIdx.x >> 3, h = blockIdx.x & 7;
  const int tid = threadIdx.x;

  for (int i = tid; i < 80 * 64; i += 160) {
    int l = i >> 6, d = i & 63;
    unsigned short qv = 0, kv = 0;
    if (l < 77) {
      size_t base = ((size_t)(b * 77 + l)) * 1536 + h * 64 + d;
      qv = qkv[base];
      kv = qkv[base + 512];
    }
    Qs[i]  = qv;
    Ksh[i] = kv;
  }
  for (int i = tid; i < 64 * 96; i += 160) {
    int d = i / 96, l = i % 96;
    unsigned short vv = 0;
    if (l < 77) vv = qkv[((size_t)(b * 77 + l)) * 1536 + 1024 + h * 64 + d];
    Vt[i] = vv;
  }
  for (int i = tid; i < 80 * 16; i += 160) {      // zero P padding cols 80..95
    int rw = i >> 4, c = i & 15;
    Ps[rw * 96 + 80 + c] = 0;
  }
  __syncthreads();

  const int wv   = tid >> 5;
  const int lane = tid & 31;
  const int half = lane >> 4, l16 = lane & 15;
  const int q0   = wv * 16;

  FragBF aq0, aq1;
  {
    const int mrow = (q0 + l16) * 64;
    const v4u* p0 = reinterpret_cast<const v4u*>(&Qs[mrow + half * 8]);
    aq0.q[0] = p0[0]; aq0.q[1] = p0[2];
    const v4u* p1 = reinterpret_cast<const v4u*>(&Qs[mrow + 32 + half * 8]);
    aq1.q[0] = p1[0]; aq1.q[1] = p1[2];
  }

  v8f sacc[5];
#pragma unroll
  for (int j = 0; j < 5; ++j) {
    FragBF bk0, bk1;
    const int krow = (j * 16 + l16) * 64;
    const v4u* p0 = reinterpret_cast<const v4u*>(&Ksh[krow + half * 8]);
    bk0.q[0] = p0[0]; bk0.q[1] = p0[2];
    const v4u* p1 = reinterpret_cast<const v4u*>(&Ksh[krow + 32 + half * 8]);
    bk1.q[0] = p1[0]; bk1.q[1] = p1[2];
    v8f z = {};
    z = __builtin_amdgcn_wmma_f32_16x16x32_bf16(false, aq0.v, false, bk0.v, (short)0, z, false, false);
    z = __builtin_amdgcn_wmma_f32_16x16x32_bf16(false, aq1.v, false, bk1.v, (short)0, z, false, false);
    sacc[j] = z;
  }

  float izrow[8];
#pragma unroll
  for (int r = 0; r < 8; ++r) {
    int qg = q0 + half * 8 + r;
    float m = -1e30f;
#pragma unroll
    for (int j = 0; j < 5; ++j) {
      int key = j * 16 + l16;
      float s = sacc[j][r] * 0.125f;
      if (key > qg || key >= 77) s = -1e9f;
      sacc[j][r] = s;
      m = fmaxf(m, s);
    }
    m = fmaxf(m, __shfl_xor(m, 1));
    m = fmaxf(m, __shfl_xor(m, 2));
    m = fmaxf(m, __shfl_xor(m, 4));
    m = fmaxf(m, __shfl_xor(m, 8));
    float zs = 0.0f;
#pragma unroll
    for (int j = 0; j < 5; ++j) {
      float p = __expf(sacc[j][r] - m);
      sacc[j][r] = p;
      zs += p;
    }
    zs += __shfl_xor(zs, 1);
    zs += __shfl_xor(zs, 2);
    zs += __shfl_xor(zs, 4);
    zs += __shfl_xor(zs, 8);
    izrow[r] = 1.0f / zs;
  }

#pragma unroll
  for (int r = 0; r < 8; ++r) {
    int prow = (q0 + half * 8 + r) * 96;
#pragma unroll
    for (int j = 0; j < 5; ++j)
      Ps[prow + j * 16 + l16] = f2bf(sacc[j][r]);
  }

  FragBF ap[3];
#pragma unroll
  for (int kc = 0; kc < 3; ++kc) {
    const int prow = (q0 + l16) * 96 + kc * 32;
    const v4u* p0 = reinterpret_cast<const v4u*>(&Ps[prow + half * 8]);
    ap[kc].q[0] = p0[0]; ap[kc].q[1] = p0[2];
  }
#pragma unroll
  for (int dj = 0; dj < 4; ++dj) {
    v8f oacc = {};
#pragma unroll
    for (int kc = 0; kc < 3; ++kc) {
      FragBF bv;
      const int vrow = (dj * 16 + l16) * 96 + kc * 32;
      const v4u* p0 = reinterpret_cast<const v4u*>(&Vt[vrow + half * 8]);
      bv.q[0] = p0[0]; bv.q[1] = p0[2];
      oacc = __builtin_amdgcn_wmma_f32_16x16x32_bf16(false, ap[kc].v, false, bv.v, (short)0, oacc, false, false);
    }
#pragma unroll
    for (int r = 0; r < 8; ++r) {
      int qg = q0 + half * 8 + r;
      if (qg < 77)
        out[((size_t)(b * 77 + qg)) * 512 + h * 64 + dj * 16 + l16] =
            f2bf(oacc[r] * izrow[r]);
    }
  }
}

// ---------------------------------------------------------------------------
// VALU attention for the L=6 aggregator (bf16 in/out).
// ---------------------------------------------------------------------------
__global__ __launch_bounds__(128)
void attn_kernel(const unsigned short* __restrict__ qkv,
                 unsigned short* __restrict__ out, int L, int causal)
{
  __shared__ float Ks[77 * 64];
  __shared__ float Vs[77 * 64];
  __shared__ float Sc[77 * 78];
  int b = blockIdx.x >> 3, h = blockIdx.x & 7;
  int tid = threadIdx.x;
  for (int i = tid; i < L * 64; i += 128) {
    int l = i >> 6, d = i & 63;
    size_t base = ((size_t)(b * L + l)) * 1536 + h * 64 + d;
    Ks[i] = bf2f(qkv[base + 512]);
    Vs[i] = bf2f(qkv[base + 1024]);
  }
  __syncthreads();
  for (int l = tid; l < L; l += 128) {
    size_t qb = ((size_t)(b * L + l)) * 1536 + h * 64;
    float q[64];
#pragma unroll
    for (int d = 0; d < 64; ++d) q[d] = bf2f(qkv[qb + d]);
    float mx = -1e30f;
    for (int k = 0; k < L; ++k) {
      float s = 0.0f;
#pragma unroll
      for (int d = 0; d < 64; ++d) s += q[d] * Ks[k * 64 + d];
      s *= 0.125f;
      if (causal && k > l) s = -1e9f;
      Sc[l * 78 + k] = s;
      mx = fmaxf(mx, s);
    }
    float Z = 0.0f;
    for (int k = 0; k < L; ++k) {
      float p = __expf(Sc[l * 78 + k] - mx);
      Sc[l * 78 + k] = p; Z += p;
    }
    float iz = 1.0f / Z;
    float acc[64];
#pragma unroll
    for (int d = 0; d < 64; ++d) acc[d] = 0.0f;
    for (int k = 0; k < L; ++k) {
      float p = Sc[l * 78 + k];
#pragma unroll
      for (int d = 0; d < 64; ++d) acc[d] += p * Vs[k * 64 + d];
    }
    size_t ob = ((size_t)(b * L + l)) * 512 + h * 64;
#pragma unroll
    for (int d = 0; d < 64; ++d) out[ob + d] = f2bf(acc[d] * iz);
  }
}

// ---------------------------------------------------------------------------
// LayerNorm D=512, one wave per row. OBF=1 -> bf16 output.
// ---------------------------------------------------------------------------
template<int OBF>
__global__ __launch_bounds__(128)
void ln512_kernel(const float* __restrict__ X, const float* __restrict__ g,
                  const float* __restrict__ b, void* __restrict__ Yv, int M)
{
  int row = blockIdx.x * 4 + (threadIdx.x >> 5);
  int lane = threadIdx.x & 31;
  if (row >= M) return;
  const float* x = X + (size_t)row * 512;
  float r[16]; float s = 0.0f;
#pragma unroll
  for (int i = 0; i < 16; ++i) { r[i] = x[lane + 32 * i]; s += r[i]; }
#pragma unroll
  for (int o = 16; o > 0; o >>= 1) s += __shfl_xor(s, o);
  float mean = s * (1.0f / 512.0f);
  float vs = 0.0f;
#pragma unroll
  for (int i = 0; i < 16; ++i) { float d = r[i] - mean; vs += d * d; }
#pragma unroll
  for (int o = 16; o > 0; o >>= 1) vs += __shfl_xor(vs, o);
  float inv = rsqrtf(vs * (1.0f / 512.0f) + 1e-5f);
#pragma unroll
  for (int i = 0; i < 16; ++i) {
    int c = lane + 32 * i;
    float v = (r[i] - mean) * inv * g[c] + b[c];
    if (OBF) ((unsigned short*)Yv)[(size_t)row * 512 + c] = f2bf(v);
    else     ((float*)Yv)[(size_t)row * 512 + c] = v;
  }
}

// ---------------------------------------------------------------------------
// Row L2-normalize + scale, D=512. OBF=1 -> bf16 output.
// ---------------------------------------------------------------------------
template<int OBF>
__global__ __launch_bounds__(128)
void l2scale_kernel(const float* __restrict__ X, void* __restrict__ Yv,
                    int M, float scale, float eps)
{
  int row = blockIdx.x * 4 + (threadIdx.x >> 5);
  int lane = threadIdx.x & 31;
  if (row >= M) return;
  const float* x = X + (size_t)row * 512;
  float r[16]; float ss = 0.0f;
#pragma unroll
  for (int i = 0; i < 16; ++i) { r[i] = x[lane + 32 * i]; ss += r[i] * r[i]; }
#pragma unroll
  for (int o = 16; o > 0; o >>= 1) ss += __shfl_xor(ss, o);
  float denom = fmaxf(sqrtf(ss), eps);
  float k = scale / denom;
#pragma unroll
  for (int i = 0; i < 16; ++i) {
    float v = r[i] * k;
    if (OBF) ((unsigned short*)Yv)[(size_t)row * 512 + lane + 32 * i] = f2bf(v);
    else     ((float*)Yv)[(size_t)row * 512 + lane + 32 * i] = v;
  }
}

// ---------------------------------------------------------------------------
// Top-k and glue kernels.
// ---------------------------------------------------------------------------
__global__ void topk_obj_kernel(const float* __restrict__ scores, int n,
                                int* __restrict__ sel, float* __restrict__ tmp)
{
  __shared__ float bv[256]; __shared__ int bi[256];
  int tid = threadIdx.x;
  for (int i = tid; i < n; i += 256) tmp[i] = scores[i];
  __syncthreads();
  for (int it = 0; it < 32; ++it) {
    float best = -1e30f; int bidx = 0x7fffffff;
    for (int i = tid; i < n; i += 256) {
      float v = tmp[i];
      if (v > best || (v == best && i < bidx)) { best = v; bidx = i; }
    }
    bv[tid] = best; bi[tid] = bidx; __syncthreads();
    for (int s = 128; s > 0; s >>= 1) {
      if (tid < s) {
        if (bv[tid + s] > bv[tid] || (bv[tid + s] == bv[tid] && bi[tid + s] < bi[tid])) {
          bv[tid] = bv[tid + s]; bi[tid] = bi[tid + s];
        }
      }
      __syncthreads();
    }
    if (tid == 0) { sel[it] = bi[0]; tmp[bi[0]] = -1e30f; }
    __syncthreads();
  }
}

__global__ __launch_bounds__(128)
void topk5_kernel(const float* __restrict__ logits, const int* __restrict__ sel,
                  int* __restrict__ tki)
{
  __shared__ float bv[128]; __shared__ int bi[128]; __shared__ int chosen[5];
  int s = blockIdx.x, tid = threadIdx.x;
  const float* row = logits + (size_t)sel[s] * 1204;
  for (int it = 0; it < 5; ++it) {
    float best = -1e30f; int bidx = 0x7fffffff;
    for (int i = tid; i < 1203; i += 128) {
      bool skip = false;
      for (int t = 0; t < it; ++t) if (chosen[t] == i) skip = true;
      if (!skip) {
        float v = row[i];
        if (v > best || (v == best && i < bidx)) { best = v; bidx = i; }
      }
    }
    bv[tid] = best; bi[tid] = bidx; __syncthreads();
    for (int st = 64; st > 0; st >>= 1) {
      if (tid < st) {
        if (bv[tid + st] > bv[tid] || (bv[tid + st] == bv[tid] && bi[tid + st] < bi[tid])) {
          bv[tid] = bv[tid + st]; bi[tid] = bi[tid + st];
        }
      }
      __syncthreads();
    }
    if (tid == 0) { chosen[it] = bi[0]; tki[s * 5 + it] = bi[0]; }
    __syncthreads();
  }
}

__global__ void gather_rows_bf_kernel(const unsigned short* __restrict__ X,
                                      const int* __restrict__ sel,
                                      unsigned short* __restrict__ Y, int ncols)
{
  int idx = blockIdx.x * 256 + threadIdx.x;
  int r = idx / ncols, c = idx % ncols;
  if (r < 32) Y[idx] = X[(size_t)sel[r] * ncols + c];
}

__global__ void eot_kernel(const int* __restrict__ eos_index, const int* __restrict__ tki,
                           int* __restrict__ eot)
{
  int r = threadIdx.x;
  if (r < 160) eot[r] = eos_index[tki[r]] + 1;
}

__global__ void build_agg_kernel(const float* __restrict__ ins_x, const float* __restrict__ ex,
                                 float* __restrict__ out)
{
  int idx = blockIdx.x * 256 + threadIdx.x;
  if (idx >= 160 * 6 * 512) return;
  int d = idx & 511, pos = (idx >> 9) % 6, r = idx / (6 * 512), s = r / 5;
  out[idx] = (pos == 0) ? ins_x[(size_t)s * 512 + d]
                        : ex[((size_t)s * 5 + (pos - 1)) * 512 + d];
}

__global__ void instok_kernel(const float* __restrict__ xagg, float* __restrict__ ins_tok)
{
  int idx = blockIdx.x * 256 + threadIdx.x;
  if (idx >= 160 * 512) return;
  int r = idx >> 9, d = idx & 511;
  ins_tok[idx] = xagg[((size_t)r * 6) * 512 + d];
}

__global__ void build_txt_kernel(const float* __restrict__ tok, const int* __restrict__ tki,
                                 const float* __restrict__ ins_tok, const float* __restrict__ pos,
                                 float* __restrict__ out)
{
  int idx = blockIdx.x * 256 + threadIdx.x;
  if (idx >= 12320 * 512) return;
  int d = idx & 511, row = idx >> 9, r = row / 77, s = row % 77;
  int c = tki[r];
  float v;
  if (s < 2)       v = tok[((size_t)c * 77 + s) * 512 + d];
  else if (s == 2) v = ins_tok[(size_t)r * 512 + d];
  else             v = tok[((size_t)c * 77 + (s - 1)) * 512 + d];
  out[idx] = v + pos[s * 512 + d];
}

__global__ void gather_eot_kernel(const float* __restrict__ t, const int* __restrict__ eot,
                                  unsigned short* __restrict__ out)
{
  int idx = blockIdx.x * 256 + threadIdx.x;
  if (idx >= 160 * 512) return;
  int r = idx >> 9, d = idx & 511;
  out[idx] = f2bf(t[((size_t)(r * 77 + eot[r])) * 512 + d]);
}

__global__ __launch_bounds__(128)
void rerank_kernel(const float* __restrict__ ins_x, const float* __restrict__ feats,
                   float* __restrict__ rr)
{
  int r = blockIdx.x * 4 + (threadIdx.x >> 5);
  int lane = threadIdx.x & 31;
  if (r >= 160) return;
  const float* a = ins_x + (size_t)(r / 5) * 512;
  const float* f = feats + (size_t)r * 512;
  float s = 0.0f;
#pragma unroll
  for (int i = 0; i < 16; ++i) s += a[lane + 32 * i] * f[lane + 32 * i];
#pragma unroll
  for (int o = 16; o > 0; o >>= 1) s += __shfl_xor(s, o);
  if (lane == 0) rr[r] = 50.0f * s;
}

__global__ void fuse_kernel(const float* __restrict__ logits, const float* __restrict__ rr,
                            const int* __restrict__ sel, const int* __restrict__ tki,
                            float* __restrict__ logits_final)
{
  int idx = threadIdx.x;
  if (idx >= 160) return;
  int s = idx / 5, row = sel[s], c = tki[idx];
  float a = logits[(size_t)row * 1204 + c];
  float b = rr[idx];
  float pa = 1.0f / (1.0f + __expf(-a));
  float pb = 1.0f / (1.0f + __expf(-b));
  float p = __powf(pa, 0.75f) * __powf(pb, 0.25f);
  p = fminf(fmaxf(p, 0.0f), 1.0f);
  float x1 = fmaxf(p, 1e-5f), x2 = fmaxf(1.0f - p, 1e-5f);
  logits_final[(size_t)row * 1204 + c] = __logf(x1 / x2);
}

// ---------------------------------------------------------------------------
// Host-side helpers.
// ---------------------------------------------------------------------------
template<int EP, int OBF>
static inline void launch_gemm(const unsigned short* A, const unsigned short* Bt,
                               const float* bias, const float* Res, void* C,
                               int M, int N, int K, hipStream_t st)
{
  dim3 g((N + 63) / 64, (M + 63) / 64);
  gemm_kernel<EP, OBF><<<g, dim3(128), 0, st>>>(A, Bt, bias, Res, C, M, N, K);
}

static inline void launch_tr(const float* W, unsigned short* Wt, int K, int N,
                             hipStream_t st)
{
  int n = K * N;
  transpose_bf16_kernel<<<(n + 255) / 256, 256, 0, st>>>(W, Wt, K, N);
}

// P: ln1_g, ln1_b, ln2_g, ln2_b, wqkv, bqkv, wo, bo, w1, b1, w2, b2 (f32)
// Wt: transposed bf16 weights {wqkvT, woT, w1T, w2T}, per-layer contiguous.
static void run_tf(const float* const* P, unsigned short* const* Wt,
                   int layers, int Bn, int L, int causal,
                   float* x, unsigned short* xn_bf, unsigned short* u_bf,
                   hipStream_t stream)
{
  const int M = Bn * L;
  unsigned short* qkv_bf  = u_bf;                        // [M,1536]
  unsigned short* attn_bf = u_bf + (size_t)M * 1536;     // [M,512]
  unsigned short* h_bf    = u_bf;                        // [M,2048]
  dim3 lnG((M + 3) / 4), lnB(128);
  for (int l = 0; l < layers; ++l) {
    const float* ln1g = P[0] + (size_t)l * 512;
    const float* ln1b = P[1] + (size_t)l * 512;
    const float* ln2g = P[2] + (size_t)l * 512;
    const float* ln2b = P[3] + (size_t)l * 512;
    const float* bqkv = P[5] + (size_t)l * 1536;
    const float* bo   = P[7] + (size_t)l * 512;
    const float* b1   = P[9] + (size_t)l * 2048;
    const float* b2   = P[11] + (size_t)l * 512;
    const unsigned short* wqkvT = Wt[0] + (size_t)l * 512 * 1536;
    const unsigned short* woT   = Wt[1] + (size_t)l * 512 * 512;
    const unsigned short* w1T   = Wt[2] + (size_t)l * 512 * 2048;
    const unsigned short* w2T   = Wt[3] + (size_t)l * 2048 * 512;

    ln512_kernel<1><<<lnG, lnB, 0, stream>>>(x, ln1g, ln1b, xn_bf, M);
    launch_gemm<0, 1>(xn_bf, wqkvT, bqkv, nullptr, qkv_bf, M, 1536, 512, stream);
    if (L == 77 && causal)
      attn77_wmma_kernel<<<dim3(Bn * 8), dim3(160), 0, stream>>>(qkv_bf, attn_bf);
    else
      attn_kernel<<<dim3(Bn * 8), dim3(128), 0, stream>>>(qkv_bf, attn_bf, L, causal);
    launch_gemm<2, 0>(attn_bf, woT, bo, x, x, M, 512, 512, stream);
    ln512_kernel<1><<<lnG, lnB, 0, stream>>>(x, ln2g, ln2b, xn_bf, M);
    launch_gemm<1, 1>(xn_bf, w1T, b1, nullptr, h_bf, M, 2048, 512, stream);
    launch_gemm<2, 0>(h_bf, w2T, b2, x, x, M, 512, 2048, stream);
  }
}

extern "C" void kernel_launch(void* const* d_in, const int* in_sizes, int n_in,
                              void* d_out, int out_size, void* d_ws, size_t ws_size,
                              hipStream_t stream)
{
  const float* input_x   = (const float*)d_in[0];
  const float* obj       = (const float*)d_in[1];
  const float* w_lin     = (const float*)d_in[3];
  const float* b_lin     = (const float*)d_in[4];
  const float* w_ins     = (const float*)d_in[5];
  const float* b_ins     = (const float*)d_in[6];
  const float* zsw       = (const float*)d_in[7];
  const float* token_seq = (const float*)d_in[8];
  const float* exf       = (const float*)d_in[9];
  const int*   eos       = (const int*)d_in[10];
  const float* aggP[12]; for (int i = 0; i < 12; ++i) aggP[i] = (const float*)d_in[11 + i];
  const float* txt_pos   = (const float*)d_in[23];
  const float* txtP[12]; for (int i = 0; i < 12; ++i) txtP[i] = (const float*)d_in[24 + i];
  const float* lnf_g     = (const float*)d_in[36];
  const float* lnf_b     = (const float*)d_in[37];
  const float* txt_proj  = (const float*)d_in[38];

  float* out_logits = (float*)d_out;
  float* out_rerank = out_logits + (size_t)1000 * 1204;
  float* out_final  = out_rerank + 160;

  char* wsB = (char*)d_ws;
  size_t ob = 0;
  auto allocF = [&](size_t n) { float* p = (float*)(wsB + ob); ob += n * 4; return p; };
  auto allocH = [&](size_t n) { unsigned short* p = (unsigned short*)(wsB + ob);
                                ob += ((n * 2 + 15) & ~(size_t)15); return p; };
  // f32 buffers
  float* xbuf    = allocF(1000 * 512);
  float* tscore  = allocF(1024);
  float* ins_x   = allocF(32 * 512);
  float* ins_tok = allocF(160 * 512);
  float* feats   = allocF(160 * 512);
  float* x_t     = allocF((size_t)12320 * 512);
  // bf16 buffers
  unsigned short* inx_bf   = allocH((size_t)1000 * 1024);
  unsigned short* xg_bf    = allocH(32 * 1024);
  unsigned short* xbuf_bf  = allocH(1000 * 512);
  unsigned short* xn_bf    = allocH((size_t)12320 * 512);
  unsigned short* u_bf     = allocH((size_t)12320 * 2048);
  unsigned short* fin_bf   = allocH(160 * 512);
  unsigned short* wlinT    = allocH((size_t)1024 * 512);
  unsigned short* winsT    = allocH((size_t)1024 * 512);
  unsigned short* zswT     = allocH((size_t)1204 * 512);
  unsigned short* tprojT   = allocH((size_t)512 * 512);
  unsigned short* aggWt[4], *txtWt[4];
  aggWt[0] = allocH((size_t)4 * 512 * 1536);
  aggWt[1] = allocH((size_t)4 * 512 * 512);
  aggWt[2] = allocH((size_t)4 * 512 * 2048);
  aggWt[3] = allocH((size_t)4 * 2048 * 512);
  txtWt[0] = allocH((size_t)12 * 512 * 1536);
  txtWt[1] = allocH((size_t)12 * 512 * 512);
  txtWt[2] = allocH((size_t)12 * 512 * 2048);
  txtWt[3] = allocH((size_t)12 * 2048 * 512);
  int* sel = (int*)(wsB + ob); ob += 128;
  int* tki = (int*)(wsB + ob); ob += 640;
  int* eot = (int*)(wsB + ob); ob += 640;
  float* xnf = (float*)u_bf;   // final-LN f32 output (u_bf free by then)

  // ---- weight/activation bf16 prep --------------------------------------
  convert_bf16_kernel<<<(1000 * 1024 + 255) / 256, 256, 0, stream>>>(
      input_x, inx_bf, 1000 * 1024);
  launch_tr(w_lin, wlinT, 1024, 512, stream);
  launch_tr(w_ins, winsT, 1024, 512, stream);
  launch_tr(zsw, zswT, 512, 1204, stream);
  launch_tr(txt_proj, tprojT, 512, 512, stream);
  for (int l = 0; l < 4; ++l) {
    launch_tr(aggP[4] + (size_t)l * 512 * 1536, aggWt[0] + (size_t)l * 512 * 1536, 512, 1536, stream);
    launch_tr(aggP[6] + (size_t)l * 512 * 512,  aggWt[1] + (size_t)l * 512 * 512,  512, 512,  stream);
    launch_tr(aggP[8] + (size_t)l * 512 * 2048, aggWt[2] + (size_t)l * 512 * 2048, 512, 2048, stream);
    launch_tr(aggP[10] + (size_t)l * 2048 * 512, aggWt[3] + (size_t)l * 2048 * 512, 2048, 512, stream);
  }
  for (int l = 0; l < 12; ++l) {
    launch_tr(txtP[4] + (size_t)l * 512 * 1536, txtWt[0] + (size_t)l * 512 * 1536, 512, 1536, stream);
    launch_tr(txtP[6] + (size_t)l * 512 * 512,  txtWt[1] + (size_t)l * 512 * 512,  512, 512,  stream);
    launch_tr(txtP[8] + (size_t)l * 512 * 2048, txtWt[2] + (size_t)l * 512 * 2048, 512, 2048, stream);
    launch_tr(txtP[10] + (size_t)l * 2048 * 512, txtWt[3] + (size_t)l * 2048 * 512, 2048, 512, stream);
  }

  // ---- selection + class logits -----------------------------------------
  topk_obj_kernel<<<1, 256, 0, stream>>>(obj, 1000, sel, tscore);
  gather_rows_bf_kernel<<<128, 256, 0, stream>>>(inx_bf, sel, xg_bf, 1024);

  launch_gemm<0, 0>(inx_bf, wlinT, b_lin, nullptr, xbuf, 1000, 512, 1024, stream);
  l2scale_kernel<1><<<(1000 + 3) / 4, 128, 0, stream>>>(xbuf, xbuf_bf, 1000, 50.0f, 1e-12f);
  launch_gemm<0, 0>(xbuf_bf, zswT, nullptr, nullptr, out_logits, 1000, 1204, 512, stream);
  (void)hipMemcpyAsync(out_final, out_logits, (size_t)1000 * 1204 * sizeof(float),
                       hipMemcpyDeviceToDevice, stream);

  launch_gemm<0, 0>(xg_bf, winsT, b_ins, nullptr, ins_x, 32, 512, 1024, stream);
  l2scale_kernel<0><<<8, 128, 0, stream>>>(ins_x, ins_x, 32, 1.0f, 1e-12f);

  topk5_kernel<<<32, 128, 0, stream>>>(out_logits, sel, tki);
  eot_kernel<<<1, 192, 0, stream>>>(eos, tki, eot);

  // ---- aggregator transformer [160, 6, 512] ------------------------------
  build_agg_kernel<<<(160 * 6 * 512) / 256, 256, 0, stream>>>(ins_x, exf, x_t);
  run_tf(aggP, aggWt, 4, 160, 6, 0, x_t, xn_bf, u_bf, stream);
  instok_kernel<<<(160 * 512) / 256, 256, 0, stream>>>(x_t, ins_tok);

  // ---- text transformer [160, 77, 512] -----------------------------------
  build_txt_kernel<<<(12320 * 512) / 256, 256, 0, stream>>>(token_seq, tki, ins_tok,
                                                            txt_pos, x_t);
  run_tf(txtP, txtWt, 12, 160, 77, 1, x_t, xn_bf, u_bf, stream);

  // ---- head --------------------------------------------------------------
  ln512_kernel<0><<<(12320 + 3) / 4, 128, 0, stream>>>(x_t, lnf_g, lnf_b, xnf, 12320);
  gather_eot_kernel<<<(160 * 512) / 256, 256, 0, stream>>>(xnf, eot, fin_bf);
  launch_gemm<0, 0>(fin_bf, tprojT, nullptr, nullptr, feats, 160, 512, 512, stream);
  l2scale_kernel<0><<<40, 128, 0, stream>>>(feats, feats, 160, 1.0f, 1e-30f);
  rerank_kernel<<<40, 128, 0, stream>>>(ins_x, feats, out_rerank);

  fuse_kernel<<<1, 192, 0, stream>>>(out_logits, out_rerank, sel, tki, out_final);
}
